// Model_18975165514625
// MI455X (gfx1250) — matrix operand order repaired and checked by reading, not compiled, not run on hardware
//
#include <hip/hip_runtime.h>
#include <hip/hip_bf16.h>
#include <stdint.h>

// ---------------------------------------------------------------------------
// MI455X (gfx1250) implementation: wave32, WMMA f32_16x16x32_f16 for all GEMMs,
// fused WMMA+online-logsumexp for InfoNCE (never materializes the 8192x8192 sim).
// Weights are converted to f16 *transposed* so both WMMA operands load as
// contiguous b128 pairs.
// ---------------------------------------------------------------------------

typedef __attribute__((ext_vector_type(16))) _Float16 v16h;
typedef __attribute__((ext_vector_type(8)))  float    v8f;

#define N_NODES 8192
#define D_IN    256
#define D_HID   512
#define NNZ_E   131072
#define BATCH   16384
#define EPS_BN  1e-5f
#define NORMEPS 1e-12f
#define LEAK    0.01f
#define INVTEMP (1.0f/0.7f)
#define LSE_MAX 1.4285715f   // |cos|/0.7 upper bound -> fixed LSE shift

static inline int cdiv(int a, int b) { return (a + b - 1) / b; }

// ------------------------------ WMMA fragment loaders ----------------------
// A fragment (16x32 f16, M rows in lanes): ISA 7.12.2 layout.
// lane L holds row L&15; halfs 0..7 = K +(L>=16?8:0)+0..7, halfs 8..15 = +16.
__device__ inline v16h load_afrag(const _Float16* A, int lda, int row0, int k0, int lane) {
  const _Float16* p = A + (size_t)(row0 + (lane & 15)) * lda + k0 + ((lane >> 4) << 3);
  v16h a;
#pragma unroll
  for (int i = 0; i < 8; ++i) a[i] = p[i];
#pragma unroll
  for (int i = 0; i < 8; ++i) a[8 + i] = p[16 + i];
  return a;
}

// B fragment (32x16) where B = Z^T and Z is row-major [N x K]:
// lane L holds column (L&15) = Z row; halfs 0..7 = K +(L>=16?8:0)+0..7, halfs 8..15 = +16.
__device__ inline v16h load_bfrag_T(const _Float16* Z, int ldz, int k0, int col0, int lane) {
  const _Float16* p = Z + (size_t)(col0 + (lane & 15)) * ldz + k0 + ((lane >> 4) << 3);
  v16h b;
#pragma unroll
  for (int i = 0; i < 8; ++i) b[i] = p[i];
#pragma unroll
  for (int i = 0; i < 8; ++i) b[8 + i] = p[16 + i];
  return b;
}

// ---------------- GEMM: C[M,N] = A(f16,[MxK]) @ BT(f16,[NxK])^T [+bias] ----
// One wave computes a 16x32 strip (two C tiles sharing the A fragment).
__global__ void k_wmma_gemm(const _Float16* __restrict__ A, const _Float16* __restrict__ BT,
                            const float* __restrict__ bias, float* __restrict__ C,
                            int M, int N, int K) {
  const int lane = threadIdx.x & 31;
  const int wave = threadIdx.x >> 5;
  const int tiles_n2 = N >> 5;                       // 32-wide column strips
  const int tiles_total = (M >> 4) * tiles_n2;
  const int tile = blockIdx.x * (blockDim.x >> 5) + wave;
  if (tile >= tiles_total) return;                   // wave-uniform exit
  const int tm = tile / tiles_n2;
  const int tn = (tile - tm * tiles_n2) << 1;        // first of two 16-col tiles
  v8f acc0 = {}, acc1 = {};
  for (int k = 0; k < K; k += 32) {
    if (k + 32 < K)
      __builtin_prefetch(A + (size_t)(tm * 16 + (lane & 15)) * K + k + 32, 0, 1);
    v16h a  = load_afrag(A, K, tm * 16, k, lane);
    v16h b0 = load_bfrag_T(BT, K, k, (tn + 0) * 16, lane);
    v16h b1 = load_bfrag_T(BT, K, k, (tn + 1) * 16, lane);
    acc0 = __builtin_amdgcn_wmma_f32_16x16x32_f16(false, a, false, b0, (short)0, acc0, false, false);
    acc1 = __builtin_amdgcn_wmma_f32_16x16x32_f16(false, a, false, b1, (short)0, acc1, false, false);
  }
  const int rbase = tm * 16 + ((lane >> 4) << 3);
  const int col0 = (tn + 0) * 16 + (lane & 15);
  const int col1 = (tn + 1) * 16 + (lane & 15);
  const float bv0 = bias ? bias[col0] : 0.f;
  const float bv1 = bias ? bias[col1] : 0.f;
#pragma unroll
  for (int v = 0; v < 8; ++v) {
    C[(size_t)(rbase + v) * N + col0] = acc0[v] + bv0;
    C[(size_t)(rbase + v) * N + col1] = acc1[v] + bv1;
  }
}

// --------------- Fused InfoNCE: row-LSE(Z1@Z2^T / T) - diag, accumulated ---
__global__ void k_info_nce(const _Float16* __restrict__ Z1, const _Float16* __restrict__ Z2,
                           float* __restrict__ acc) {
  const int K = D_IN;
  const int lane = threadIdx.x & 31;
  const int wave = threadIdx.x >> 5;
  const int ti = blockIdx.x * (blockDim.x >> 5) + wave;   // row tile (N/16 total)
  if (ti >= (N_NODES >> 4)) return;
  const int hi = lane >> 4;
  const int colL = lane & 15;

  v16h a[8];
#pragma unroll
  for (int kk = 0; kk < 8; ++kk) a[kk] = load_afrag(Z1, K, ti * 16, kk * 32, lane);

  float s_run[8];
#pragma unroll
  for (int v = 0; v < 8; ++v) s_run[v] = 0.f;
  float diag = 0.f;

  for (int tj = 0; tj < (N_NODES >> 4); ++tj) {
    if (tj + 1 < (N_NODES >> 4))
      __builtin_prefetch(Z2 + (size_t)(tj + 1) * 16 * K + (size_t)colL * K, 0, 1);
    v8f c = {};
#pragma unroll
    for (int kk = 0; kk < 8; ++kk) {
      v16h b = load_bfrag_T(Z2, K, kk * 32, tj * 16, lane);
      c = __builtin_amdgcn_wmma_f32_16x16x32_f16(false, a[kk], false, b, (short)0, c, false, false);
    }
#pragma unroll
    for (int v = 0; v < 8; ++v) {
      const float x = c[v] * INVTEMP;
      const int row_in = v + (hi << 3);
      if (tj == ti && colL == row_in) diag = x;    // this lane owns sim[r][r]
      s_run[v] += __expf(x - LSE_MAX);             // fixed shift: |x| <= 1/0.7
    }
  }
  // Per-row sum over the 16 columns held by this lane group, then LSE - diag.
  float contrib = 0.f;
  if ((hi == 0 && colL < 8) || (hi == 1 && colL >= 8)) {
    const int v = colL & 7;
    float s = s_run[v];
#pragma unroll
    for (int m = 1; m <= 8; m <<= 1) s += __shfl_xor(s, m, 32);
    contrib = LSE_MAX + __logf(s) - diag;
  }
#pragma unroll
  for (int m = 1; m <= 16; m <<= 1) contrib += __shfl_xor(contrib, m, 32);
  if (lane == 0) atomicAdd(acc, contrib);
}

// ------------------------------ elementwise / utility ----------------------
__global__ void k_cvt_f16(const float* __restrict__ x, _Float16* __restrict__ y, size_t n) {
  for (size_t i = (size_t)blockIdx.x * blockDim.x + threadIdx.x; i < n;
       i += (size_t)gridDim.x * blockDim.x)
    y[i] = (_Float16)x[i];
}

// Transposing f32->f16 convert: W[K x N] -> WT[N x K].
__global__ void k_cvt_f16_T(const float* __restrict__ W, _Float16* __restrict__ WT,
                            int K, int N) {
  const size_t total = (size_t)K * N;
  for (size_t i = (size_t)blockIdx.x * blockDim.x + threadIdx.x; i < total;
       i += (size_t)gridDim.x * blockDim.x) {
    const int k = (int)(i / N), n = (int)(i % N);
    WT[(size_t)n * K + k] = (_Float16)W[i];
  }
}

__global__ void k_count(const int* __restrict__ idx, float* __restrict__ cnt, int nnz) {
  int i = blockIdx.x * blockDim.x + threadIdx.x;
  if (i < nnz) atomicAdd(&cnt[idx[i]], 1.0f);
}

__global__ void k_scatter_add(const float* __restrict__ src, const int* __restrict__ gidx,
                              const int* __restrict__ sidx, float* __restrict__ dst,
                              int nnz, int D, int dshift) {
  const size_t total = (size_t)nnz * D;
  for (size_t i = (size_t)blockIdx.x * blockDim.x + threadIdx.x; i < total;
       i += (size_t)gridDim.x * blockDim.x) {
    const int e = (int)(i >> dshift), d = (int)(i & (D - 1));
    atomicAdd(&dst[(size_t)sidx[e] * D + d], src[(size_t)gidx[e] * D + d]);
  }
}

__global__ void k_scale_rows(float* __restrict__ X, const float* __restrict__ cnt,
                             const float* __restrict__ bias, int N, int D, int dshift) {
  const size_t total = (size_t)N * D;
  for (size_t i = (size_t)blockIdx.x * blockDim.x + threadIdx.x; i < total;
       i += (size_t)gridDim.x * blockDim.x) {
    const int r = (int)(i >> dshift), d = (int)(i & (D - 1));
    const float c = cnt[r];
    float v = X[i] * (c > 0.f ? 1.f / c : 0.f);
    if (bias) v += bias[d];
    X[i] = v;
  }
}

__global__ void k_colstats(const float* __restrict__ X, int M, int D,
                           float* __restrict__ mean, float* __restrict__ var) {
  const int d = blockIdx.x * blockDim.x + threadIdx.x;
  if (d >= D) return;
  float s = 0.f, s2 = 0.f;
  for (int m = 0; m < M; ++m) { const float x = X[(size_t)m * D + d]; s += x; s2 += x * x; }
  const float mu = s / M;
  mean[d] = mu;
  var[d] = s2 / M - mu * mu;
}

__global__ void k_bn_apply(const float* __restrict__ X, const float* __restrict__ mean,
                           const float* __restrict__ var, const float* __restrict__ g,
                           const float* __restrict__ b, float* __restrict__ Y,
                           int M, int D, int leaky) {
  const size_t total = (size_t)M * D;
  for (size_t i = (size_t)blockIdx.x * blockDim.x + threadIdx.x; i < total;
       i += (size_t)gridDim.x * blockDim.x) {
    const int d = (int)(i % D);
    float y = (X[i] - mean[d]) * rsqrtf(var[d] + EPS_BN) * g[d] + b[d];
    if (leaky) y = y > 0.f ? y : LEAK * y;
    Y[i] = y;
  }
}

__global__ void k_rownorm16(const float* __restrict__ Z, _Float16* __restrict__ O, int N, int D) {
  const int wave = (int)(((size_t)blockIdx.x * blockDim.x + threadIdx.x) >> 5);
  const int lane = threadIdx.x & 31;
  if (wave >= N) return;
  const float* row = Z + (size_t)wave * D;
  float s = 0.f;
  for (int j = lane; j < D; j += 32) { const float v = row[j]; s += v * v; }
#pragma unroll
  for (int m = 1; m <= 16; m <<= 1) s += __shfl_xor(s, m, 32);
  const float inv = 1.f / fmaxf(sqrtf(s), NORMEPS);
  for (int j = lane; j < D; j += 32) O[(size_t)wave * D + j] = (_Float16)(row[j] * inv);
}

// Row attention: channel0=c0 (conv_3), channel1=c1 (conv_2); + residual x.
__global__ void k_row_att(const float* __restrict__ c0, const float* __restrict__ c1,
                          const float* __restrict__ xres,
                          const float* __restrict__ fc1W, const float* __restrict__ fc1b,
                          const float* __restrict__ fc2W, const float* __restrict__ fc2b,
                          const float* __restrict__ crW, const float* __restrict__ crb,
                          float* __restrict__ out, int N, int D) {
  const int wave = (int)(((size_t)blockIdx.x * blockDim.x + threadIdx.x) >> 5);
  const int lane = threadIdx.x & 31;
  if (wave >= N) return;
  const size_t off = (size_t)wave * D;
  float s0 = 0.f, s1 = 0.f;
  for (int j = lane; j < D; j += 32) { s0 += c0[off + j]; s1 += c1[off + j]; }
#pragma unroll
  for (int m = 1; m <= 16; m <<= 1) { s0 += __shfl_xor(s0, m, 32); s1 += __shfl_xor(s1, m, 32); }
  const float p0 = s0 / D, p1 = s1 / D;
  float a = p0 * fc1W[0] + p1 * fc1W[1] + fc1b[0];
  a = fmaxf(a, 0.f);
  const float a0 = 1.f / (1.f + __expf(-(a * fc2W[0] + fc2b[0])));
  const float a1 = 1.f / (1.f + __expf(-(a * fc2W[1] + fc2b[1])));
  const float w0 = crW[0], w1 = crW[1], cb = crb[0];
  for (int j = lane; j < D; j += 32)
    out[off + j] = w0 * (c0[off + j] * a0) + w1 * (c1[off + j] * a1) + cb + xres[off + j];
}

__global__ void k_gather_concat(const float* __restrict__ P, const float* __restrict__ M_,
                                const int* __restrict__ i0, const int* __restrict__ i1,
                                float* __restrict__ F, int B, int Dp, int Dm) {
  const size_t total = (size_t)B * (Dp + Dm);
  for (size_t i = (size_t)blockIdx.x * blockDim.x + threadIdx.x; i < total;
       i += (size_t)gridDim.x * blockDim.x) {
    const int b = (int)(i / (Dp + Dm)), d = (int)(i % (Dp + Dm));
    F[i] = (d < Dp) ? P[(size_t)i0[b] * Dp + d] : M_[(size_t)i1[b] * Dm + (d - Dp)];
  }
}

__global__ void k_head(const float* __restrict__ H, const float* __restrict__ W,
                       const float* __restrict__ b, float* __restrict__ out, int M, int K) {
  const int r = blockIdx.x * blockDim.x + threadIdx.x;
  if (r >= M) return;
  float s = b[0];
  for (int k = 0; k < K; ++k) s += H[(size_t)r * K + k] * W[k];
  out[r] = s;
}

__global__ void k_loss(float* __restrict__ out_loss, const float* __restrict__ accs,
                       const float* __restrict__ p12, const float* __restrict__ m13) {
  if (threadIdx.x == 0 && blockIdx.x == 0) {
    const float lp = 0.5f * accs[0] / (float)N_NODES;
    const float lm = 0.5f * accs[1] / (float)N_NODES;
    out_loss[0] = __expf(-p12[0]) * lp + p12[0] + __expf(-m13[0]) * lm + m13[0];
  }
}

// ------------------------------ input index map ----------------------------
// Top level: setup_inputs() insertion order; `params` expanded as a JAX pytree
// (dict keys sorted ASCII, capitals before lowercase).
enum {
  IN_XP = 0, IN_XM = 1,
  IN_ATTM_CRW = 2, IN_ATTM_CRB, IN_ATTM_FC1W, IN_ATTM_FC1B, IN_ATTM_FC2W, IN_ATTM_FC2B,
  IN_ATTP_CRW = 8, IN_ATTP_CRB, IN_ATTP_FC1W, IN_ATTP_FC1B, IN_ATTP_FC2W, IN_ATTP_FC2B,
  IN_ENC_W1 = 14, IN_ENC_W2, IN_ENC_W3, IN_ENC_B1, IN_ENC_B2, IN_ENC_B3,
  IN_ENC_BN1B = 20, IN_ENC_BN1G, IN_ENC_BN2B, IN_ENC_BN2G,
  IN_M1_W1 = 24, IN_M1_W2, IN_M1_B1, IN_M1_B2, IN_M1_BNB, IN_M1_BNG,
  IN_M13 = 30,
  IN_M2_W1 = 31, IN_M2_W2, IN_M2_B1, IN_M2_B2, IN_M2_BNB, IN_M2_BNG,
  IN_P1_W1 = 37, IN_P1_W2, IN_P1_B1, IN_P1_B2, IN_P1_BNB, IN_P1_BNG,
  IN_P12 = 43,
  IN_P2_W1 = 44, IN_P2_W2, IN_P2_B1, IN_P2_B2, IN_P2_BNB, IN_P2_BNG,
  IN_HEP = 50, IN_HEM = 51, IN_INDEX = 52
};

extern "C" void kernel_launch(void* const* d_in, const int* in_sizes, int n_in,
                              void* d_out, int out_size, void* d_ws, size_t ws_size,
                              hipStream_t stream) {
  (void)in_sizes; (void)n_in; (void)out_size; (void)ws_size;
  const float* xp = (const float*)d_in[IN_XP];
  const float* xm = (const float*)d_in[IN_XM];
  const int* hep = (const int*)d_in[IN_HEP];
  const int* hem = (const int*)d_in[IN_HEM];
  const int* idx = (const int*)d_in[IN_INDEX];
  float* out = (float*)d_out;

  // ---- workspace carve (bump allocator, 256B aligned) ----
  uint8_t* wp = (uint8_t*)d_ws;
  auto alloc = [&](size_t bytes) -> void* {
    void* p = wp; wp += (bytes + 255) & ~(size_t)255; return p;
  };
  float* bufA = (float*)alloc((size_t)N_NODES * D_HID * 4);
  float* bufB = (float*)alloc((size_t)N_NODES * D_HID * 4);
  float* bufC = (float*)alloc((size_t)N_NODES * D_HID * 4);
  float* p2o  = (float*)alloc((size_t)N_NODES * D_IN * 4);
  float* p3o  = (float*)alloc((size_t)N_NODES * D_IN * 4);
  float* m2o  = (float*)alloc((size_t)N_NODES * D_IN * 4);
  float* m3o  = (float*)alloc((size_t)N_NODES * D_IN * 4);
  float* prot = (float*)alloc((size_t)N_NODES * D_IN * 4);
  float* met  = (float*)alloc((size_t)N_NODES * D_IN * 4);
  float* feat = (float*)alloc((size_t)BATCH * 512 * 4);
  float* h1a  = (float*)alloc((size_t)BATCH * 128 * 4);
  float* h1b  = (float*)alloc((size_t)BATCH * 128 * 4);
  float* h2a  = (float*)alloc((size_t)BATCH * 64 * 4);
  float* h2b  = (float*)alloc((size_t)BATCH * 64 * 4);
  _Float16* xp16 = (_Float16*)alloc((size_t)N_NODES * D_IN * 2);
  _Float16* xm16 = (_Float16*)alloc((size_t)N_NODES * D_IN * 2);
  _Float16* ah16 = (_Float16*)alloc((size_t)BATCH * 512 * 2);   // activation f16 scratch
  _Float16* wh16 = (_Float16*)alloc((size_t)512 * 512 * 2);     // transposed weight f16 scratch
  _Float16* zA = (_Float16*)alloc((size_t)N_NODES * D_IN * 2);
  _Float16* zB = (_Float16*)alloc((size_t)N_NODES * D_IN * 2);
  float* degn = (float*)alloc(N_NODES * 4);
  float* dege = (float*)alloc(N_NODES * 4);
  float* meanb = (float*)alloc(512 * 4);
  float* varb  = (float*)alloc(512 * 4);
  float* accs  = (float*)alloc(2 * 4);

  auto cvt = [&](const float* s, _Float16* d, size_t n) {
    k_cvt_f16<<<(int)((n + 255) / 256), 256, 0, stream>>>(s, d, n);
  };
  auto cvtT = [&](const float* W, _Float16* WT, int K, int N) {
    k_cvt_f16_T<<<cdiv(K * N, 256), 256, 0, stream>>>(W, WT, K, N);
  };
  auto gemm = [&](const _Float16* A, const _Float16* BT, const float* bias, float* C,
                  int M, int N, int K) {
    const int tiles = (M / 16) * (N / 32);
    k_wmma_gemm<<<cdiv(tiles, 8), 256, 0, stream>>>(A, BT, bias, C, M, N, K);
  };
  auto zero = [&](void* p, size_t bytes) { (void)hipMemsetAsync(p, 0, bytes, stream); };

  // convert raw node features once
  cvt(xp, xp16, (size_t)N_NODES * D_IN);
  cvt(xm, xm16, (size_t)N_NODES * D_IN);

  // ---- one full HGNN: hconv -> BN -> hconv ----
  auto hgnn = [&](const _Float16* x16, const int* nidx, const int* eidx,
                  const float* W1, const float* b1, const float* bng, const float* bnb,
                  const float* W2, const float* b2, float* outp) {
    const size_t elems5 = (size_t)N_NODES * D_HID, elems2 = (size_t)N_NODES * D_IN;
    const int sgrid5 = (int)(((size_t)NNZ_E * D_HID + 255) / 256);
    const int sgrid2 = (int)(((size_t)NNZ_E * D_IN + 255) / 256);
    // degrees
    zero(degn, N_NODES * 4); zero(dege, N_NODES * 4);
    k_count<<<NNZ_E / 256, 256, 0, stream>>>(nidx, degn, NNZ_E);
    k_count<<<NNZ_E / 256, 256, 0, stream>>>(eidx, dege, NNZ_E);
    // xw = x @ W1  [N,512]
    cvtT(W1, wh16, D_IN, D_HID);
    gemm(x16, wh16, nullptr, bufA, N_NODES, D_HID, D_IN);
    // ef = Be^-1 * segsum_e(xw[node])
    zero(bufB, elems5 * 4);
    k_scatter_add<<<sgrid5, 256, 0, stream>>>(bufA, nidx, eidx, bufB, NNZ_E, D_HID, 9);
    k_scale_rows<<<(int)((elems5 + 255) / 256), 256, 0, stream>>>(bufB, dege, nullptr, N_NODES, D_HID, 9);
    // h = Dn^-1 * segsum_n(ef[edge]) + b1
    zero(bufC, elems5 * 4);
    k_scatter_add<<<sgrid5, 256, 0, stream>>>(bufB, eidx, nidx, bufC, NNZ_E, D_HID, 9);
    k_scale_rows<<<(int)((elems5 + 255) / 256), 256, 0, stream>>>(bufC, degn, b1, N_NODES, D_HID, 9);
    // batchnorm
    k_colstats<<<2, 256, 0, stream>>>(bufC, N_NODES, D_HID, meanb, varb);
    k_bn_apply<<<(int)((elems5 + 255) / 256), 256, 0, stream>>>(bufC, meanb, varb, bng, bnb,
                                                                bufA, N_NODES, D_HID, 0);
    // hw = hbn @ W2  [N,256]
    cvt(bufA, ah16, elems5);
    cvtT(W2, wh16, D_HID, D_IN);
    gemm(ah16, wh16, nullptr, bufB, N_NODES, D_IN, D_HID);
    // second hconv aggregation
    zero(bufC, elems2 * 4);
    k_scatter_add<<<sgrid2, 256, 0, stream>>>(bufB, nidx, eidx, bufC, NNZ_E, D_IN, 8);
    k_scale_rows<<<(int)((elems2 + 255) / 256), 256, 0, stream>>>(bufC, dege, nullptr, N_NODES, D_IN, 8);
    zero(outp, elems2 * 4);
    k_scatter_add<<<sgrid2, 256, 0, stream>>>(bufC, eidx, nidx, outp, NNZ_E, D_IN, 8);
    k_scale_rows<<<(int)((elems2 + 255) / 256), 256, 0, stream>>>(outp, degn, b2, N_NODES, D_IN, 8);
  };

  // he[g] layout: [2, NNZ] with row0 = node idx, row1 = edge idx
  hgnn(xp16, hep + 0 * NNZ_E, hep + 1 * NNZ_E,
       (const float*)d_in[IN_P1_W1], (const float*)d_in[IN_P1_B1],
       (const float*)d_in[IN_P1_BNG], (const float*)d_in[IN_P1_BNB],
       (const float*)d_in[IN_P1_W2], (const float*)d_in[IN_P1_B2], p2o);
  hgnn(xp16, hep + 2 * NNZ_E, hep + 3 * NNZ_E,
       (const float*)d_in[IN_P2_W1], (const float*)d_in[IN_P2_B1],
       (const float*)d_in[IN_P2_BNG], (const float*)d_in[IN_P2_BNB],
       (const float*)d_in[IN_P2_W2], (const float*)d_in[IN_P2_B2], p3o);
  hgnn(xm16, hem + 0 * NNZ_E, hem + 1 * NNZ_E,
       (const float*)d_in[IN_M1_W1], (const float*)d_in[IN_M1_B1],
       (const float*)d_in[IN_M1_BNG], (const float*)d_in[IN_M1_BNB],
       (const float*)d_in[IN_M1_W2], (const float*)d_in[IN_M1_B2], m2o);
  hgnn(xm16, hem + 2 * NNZ_E, hem + 3 * NNZ_E,
       (const float*)d_in[IN_M2_W1], (const float*)d_in[IN_M2_B1],
       (const float*)d_in[IN_M2_BNG], (const float*)d_in[IN_M2_BNB],
       (const float*)d_in[IN_M2_W2], (const float*)d_in[IN_M2_B2], m3o);

  // ---- InfoNCE losses (fused WMMA + online LSE) ----
  zero(accs, 2 * 4);
  const int nwgrid = cdiv(N_NODES, 8);   // 8 rows per 256-thread block (wave per row)
  k_rownorm16<<<nwgrid, 256, 0, stream>>>(p2o, zA, N_NODES, D_IN);
  k_rownorm16<<<nwgrid, 256, 0, stream>>>(p3o, zB, N_NODES, D_IN);
  k_info_nce<<<cdiv(N_NODES / 16, 8), 256, 0, stream>>>(zA, zB, &accs[0]);
  k_info_nce<<<cdiv(N_NODES / 16, 8), 256, 0, stream>>>(zB, zA, &accs[0]);
  k_rownorm16<<<nwgrid, 256, 0, stream>>>(m2o, zA, N_NODES, D_IN);
  k_rownorm16<<<nwgrid, 256, 0, stream>>>(m3o, zB, N_NODES, D_IN);
  k_info_nce<<<cdiv(N_NODES / 16, 8), 256, 0, stream>>>(zA, zB, &accs[1]);
  k_info_nce<<<cdiv(N_NODES / 16, 8), 256, 0, stream>>>(zB, zA, &accs[1]);

  // ---- row attention + residual ----
  k_row_att<<<nwgrid, 256, 0, stream>>>(p3o, p2o, xp,
      (const float*)d_in[IN_ATTP_FC1W], (const float*)d_in[IN_ATTP_FC1B],
      (const float*)d_in[IN_ATTP_FC2W], (const float*)d_in[IN_ATTP_FC2B],
      (const float*)d_in[IN_ATTP_CRW], (const float*)d_in[IN_ATTP_CRB],
      prot, N_NODES, D_IN);
  k_row_att<<<nwgrid, 256, 0, stream>>>(m3o, m2o, xm,
      (const float*)d_in[IN_ATTM_FC1W], (const float*)d_in[IN_ATTM_FC1B],
      (const float*)d_in[IN_ATTM_FC2W], (const float*)d_in[IN_ATTM_FC2B],
      (const float*)d_in[IN_ATTM_CRW], (const float*)d_in[IN_ATTM_CRB],
      met, N_NODES, D_IN);

  // ---- encoder MLP over gathered pairs ----
  k_gather_concat<<<(int)(((size_t)BATCH * 512 + 255) / 256), 256, 0, stream>>>(
      prot, met, idx, idx + BATCH, feat, BATCH, D_IN, D_IN);
  cvt(feat, ah16, (size_t)BATCH * 512);
  cvtT((const float*)d_in[IN_ENC_W1], wh16, 512, 128);
  gemm(ah16, wh16, (const float*)d_in[IN_ENC_B1], h1a, BATCH, 128, 512);
  k_colstats<<<1, 256, 0, stream>>>(h1a, BATCH, 128, meanb, varb);
  k_bn_apply<<<(int)(((size_t)BATCH * 128 + 255) / 256), 256, 0, stream>>>(
      h1a, meanb, varb, (const float*)d_in[IN_ENC_BN1G], (const float*)d_in[IN_ENC_BN1B],
      h1b, BATCH, 128, 1);
  cvt(h1b, ah16, (size_t)BATCH * 128);
  cvtT((const float*)d_in[IN_ENC_W2], wh16, 128, 64);
  gemm(ah16, wh16, (const float*)d_in[IN_ENC_B2], h2a, BATCH, 64, 128);
  k_colstats<<<1, 256, 0, stream>>>(h2a, BATCH, 64, meanb, varb);
  k_bn_apply<<<(int)(((size_t)BATCH * 64 + 255) / 256), 256, 0, stream>>>(
      h2a, meanb, varb, (const float*)d_in[IN_ENC_BN2G], (const float*)d_in[IN_ENC_BN2B],
      h2b, BATCH, 64, 1);
  k_head<<<cdiv(BATCH, 256), 256, 0, stream>>>(h2b, (const float*)d_in[IN_ENC_W3],
                                               (const float*)d_in[IN_ENC_B3], out, BATCH, 64);

  // ---- final scalar loss -> d_out[16384] ----
  k_loss<<<1, 32, 0, stream>>>(out + BATCH, accs,
                               (const float*)d_in[IN_P12], (const float*)d_in[IN_M13]);
}
